// DynCCNN_5299989643582
// MI455X (gfx1250) — matrix operand order repaired and checked by reading, not compiled, not run on hardware
//
#include <hip/hip_runtime.h>

typedef __attribute__((ext_vector_type(16))) _Float16 v16h;
typedef __attribute__((ext_vector_type(8)))  float    v8f;
typedef __attribute__((ext_vector_type(2)))  float    v2f;
typedef __attribute__((ext_vector_type(4)))  int      i4;

// Async global->LDS DMA path (gfx1250 GLOBAL_LOAD_ASYNC_TO_LDS_B128, ASYNCcnt)
#if defined(__gfx1250__) && __has_builtin(__builtin_amdgcn_global_load_async_to_lds_b128) && __has_builtin(__builtin_amdgcn_s_wait_asynccnt)
#define USE_ASYNC_LDS 1
#else
#define USE_ASYNC_LDS 0
#endif

// Problem constants
constexpr int BATCH = 4096;
constexpr int CHANS = 8;
constexpr int TLEN  = 512;
constexpr int NBIN  = 216;
constexpr int NF    = 432;      // real+imag features
constexpr int FCH   = 16;       // F
constexpr int K4    = 4;
constexpr int OUTW  = 405;
constexpr int NCLS  = 4;
constexpr int NFFT  = 1747;
constexpr float TEMP = 30.0f;
constexpr float EPS  = 1e-5f;

// ---------------------------------------------------------------------------
// Kernel 1: build the DFT coefficient table (K-major: tab[n][k], n=0..431,
// k=0..511).  col n<216 -> cos(2*pi*(24+n)*k/1747)/512 ; n>=216 -> -sin(...)/512
// ---------------------------------------------------------------------------
__global__ void __launch_bounds__(256) k_build_tab(_Float16* __restrict__ tab)
{
    int gid = blockIdx.x * 256 + threadIdx.x;
    if (gid >= NF * TLEN) return;
    int n = gid >> 9;          // / 512
    int k = gid & 511;
    int j = (n < NBIN) ? (24 + n) : (24 + n - NBIN);
    int m = (j * k) % NFFT;    // exact integer angle reduction
    float ang = 6.28318530717958647692f * (float)m / (float)NFFT;
    float v = ((n < NBIN) ? cosf(ang) : -sinf(ang)) * (1.0f / 512.0f);
    tab[n * TLEN + k] = (_Float16)v;
}

// ---------------------------------------------------------------------------
// Kernel 2: DFT as GEMM.  Xf(32768 x 432) = x(32768 x 512) * tab^T.
// One wave per 16-row M-tile; A fragments (f16, K=512 -> 16 steps of 32) held
// in registers.  B fragments streamed through a 3-deep rolling group buffer
// (4 frags/group) so ~8 b128 loads stay in flight ahead of the WMMAs.
// ---------------------------------------------------------------------------
__global__ void __launch_bounds__(256) k_dft_gemm(
    const float* __restrict__ x, const _Float16* __restrict__ tab,
    float* __restrict__ Xf)
{
    const int tid  = threadIdx.x;
    const int lane = tid & 31;
    const int wave = tid >> 5;
    const int mtb  = blockIdx.x * 128 + wave * 16;   // this wave's M-tile base
    const int half_ = lane >> 4;                     // 0 or 1
    const int nloc  = lane & 15;
    const float* xr = x + (size_t)(mtb + nloc) * TLEN;

    // A fragments: 16x32 f16 layout -- lane half selects contiguous K run {kb..kb+15}
    v16h A[16];
#pragma unroll
    for (int s = 0; s < 16; ++s) {
        const int kb = half_ * 16 + s * 32;
        float4 f0 = *(const float4*)(xr + kb);
        float4 f1 = *(const float4*)(xr + kb + 4);
        float4 f2 = *(const float4*)(xr + kb + 8);
        float4 f3 = *(const float4*)(xr + kb + 12);
        v16h a;
        a[0]=(_Float16)f0.x; a[1]=(_Float16)f0.y; a[2]=(_Float16)f0.z; a[3]=(_Float16)f0.w;
        a[4]=(_Float16)f1.x; a[5]=(_Float16)f1.y; a[6]=(_Float16)f1.z; a[7]=(_Float16)f1.w;
        a[8]=(_Float16)f2.x; a[9]=(_Float16)f2.y; a[10]=(_Float16)f2.z; a[11]=(_Float16)f2.w;
        a[12]=(_Float16)f3.x; a[13]=(_Float16)f3.y; a[14]=(_Float16)f3.z; a[15]=(_Float16)f3.w;
        A[s] = a;
    }

    for (int nt = 0; nt < 27; ++nt) {
        const _Float16* bp = tab + (size_t)(nt * 16 + nloc) * TLEN + half_ * 16;
        if (nt < 26) __builtin_prefetch(bp + 16 * TLEN, 0, 0);  // global_prefetch_b8

        v8f acc = {};
        v16h bq[3][4];                 // rolling group buffer: 2 groups in flight
#pragma unroll
        for (int s = 0; s < 4; ++s) bq[0][s] = *(const v16h*)(bp + s * 32);
#pragma unroll
        for (int s = 0; s < 4; ++s) bq[1][s] = *(const v16h*)(bp + (4 + s) * 32);
#pragma unroll
        for (int g = 0; g < 4; ++g) {
            if (g < 2) {
#pragma unroll
                for (int s = 0; s < 4; ++s)
                    bq[(g + 2) % 3][s] = *(const v16h*)(bp + (4 * (g + 2) + s) * 32);
            }
#pragma unroll
            for (int s = 0; s < 4; ++s)
                acc = __builtin_amdgcn_wmma_f32_16x16x32_f16(
                    false, A[4 * g + s], false, bq[g % 3][s], (short)0, acc, false, false);
        }

        const int t = nt * 16 + nloc;
#pragma unroll
        for (int r = 0; r < 8; ++r)
            Xf[(size_t)(mtb + r + 8 * half_) * NF + t] = acc[r];
    }
}

// ---------------------------------------------------------------------------
// Kernel 3: fused attention-1 + dynamic conv1 (8->16 channel mix) + per-block
// BN1 partial stats.  One block (8 waves) per sample.
// wmma: v_wmma_f32_16x16x4_f32 (K=8 in 2 steps); X[b] staged via async LDS DMA.
// ---------------------------------------------------------------------------
__global__ void __launch_bounds__(256) k_conv1(
    const float* __restrict__ Xf, const float* __restrict__ w1k,
    const float* __restrict__ a1w1, const float* __restrict__ a1b1,
    const float* __restrict__ a1w2, const float* __restrict__ a1b2,
    float* __restrict__ out1, float* __restrict__ bs1)
{
    __shared__ float Xs[CHANS * NF];     // 3456 f32
    __shared__ float w1s[FCH * CHANS];   // 128
    __shared__ float attS[K4];
    __shared__ float outS[FCH * NF];     // 6912 f32
    __shared__ float redA[256];
    __shared__ float redB[256];

    const int tid  = threadIdx.x;
    const int b    = blockIdx.x;
    const int lane = tid & 31;
    const int wave = tid >> 5;
    const int half_ = lane >> 4;
    const int nloc  = lane & 15;

    // stage X[b] into LDS
#if USE_ASYNC_LDS
    {
        const float* gsrc = Xf + (size_t)b * (CHANS * NF);
        for (int i = tid; i < (CHANS * NF) / 4; i += 256)   // 864 x 16B chunks
            __builtin_amdgcn_global_load_async_to_lds_b128(
                (i4*)(gsrc + i * 4), (i4*)(Xs + i * 4), 0, 0);
        __builtin_amdgcn_s_wait_asynccnt(0);
    }
#else
    for (int i = tid; i < CHANS * NF; i += 256)
        Xs[i] = Xf[(size_t)b * (CHANS * NF) + i];
#endif
    __syncthreads();

    // p = mean over 8x432
    float ps = 0.f;
    for (int i = tid; i < CHANS * NF; i += 256) ps += Xs[i];
    redA[tid] = ps;
    __syncthreads();
    for (int off = 128; off > 0; off >>= 1) {
        if (tid < off) redA[tid] += redA[tid + off];
        __syncthreads();
    }
    if (tid == 0) {
        float p = redA[0] / (float)(CHANS * NF);
        float h = fmaxf(p * a1w1[0] + a1b1[0], 0.f);
        float lg[K4]; float mx = -1e30f;
        for (int k = 0; k < K4; ++k) {
            lg[k] = (h * a1w2[k] + a1b2[k]) * (1.0f / TEMP);
            mx = fmaxf(mx, lg[k]);
        }
        float ssum = 0.f;
        for (int k = 0; k < K4; ++k) { lg[k] = __expf(lg[k] - mx); ssum += lg[k]; }
        for (int k = 0; k < K4; ++k) attS[k] = lg[k] / ssum;
    }
    __syncthreads();
    if (tid < FCH * CHANS) {            // w[b,o,c] = sum_k att[k]*w1k[k,o,c]
        float acc = 0.f;
        for (int k = 0; k < K4; ++k) acc += attS[k] * w1k[k * (FCH * CHANS) + tid];
        w1s[tid] = acc;
    }
    __syncthreads();

    // per-sample conv1 via f32 WMMA: M=16(out ch), N=432(time, 27 tiles), K=8(in ch)
    for (int tile = wave; tile < 27; tile += 8) {
        const int tb = tile * 16;
        v8f acc = {};
#pragma unroll
        for (int s = 0; s < 2; ++s) {
            const int cb = 4 * s + half_ * 2;   // K pair for this lane half
            v2f a, bb;
            a[0]  = w1s[nloc * CHANS + cb];
            a[1]  = w1s[nloc * CHANS + cb + 1];
            bb[0] = Xs[(cb)     * NF + tb + nloc];
            bb[1] = Xs[(cb + 1) * NF + tb + nloc];
            acc = __builtin_amdgcn_wmma_f32_16x16x4_f32(
                false, a, false, bb, (short)0, acc, false, false);
        }
#pragma unroll
        for (int r = 0; r < 8; ++r) {
            const int o = r + 8 * half_;
            const int t = tb + nloc;
            out1[(size_t)b * (FCH * NF) + o * NF + t] = acc[r];
            outS[o * NF + t] = acc[r];
        }
    }
    __syncthreads();

    // deterministic per-block BN1 stats (sum, sumsq) per channel
    const int c = tid >> 4, j = tid & 15;
    float s0 = 0.f, q0 = 0.f;
    for (int t = j; t < NF; t += 16) {
        float v = outS[c * NF + t];
        s0 += v; q0 += v * v;
    }
    redA[tid] = s0; redB[tid] = q0;
    __syncthreads();
    for (int off = 8; off > 0; off >>= 1) {
        if (j < off) { redA[tid] += redA[tid + off]; redB[tid] += redB[tid + off]; }
        __syncthreads();
    }
    if (j == 0) {
        bs1[b * 32 + c]      = redA[tid];
        bs1[b * 32 + 16 + c] = redB[tid];
    }
}

// ---------------------------------------------------------------------------
// Kernel 4/7: reduce per-block BN stats -> scale/shift  (1 block)
// ---------------------------------------------------------------------------
__global__ void __launch_bounds__(256) k_bn_reduce(
    const float* __restrict__ bs, const float* __restrict__ g,
    const float* __restrict__ bb, float* __restrict__ st, float invN)
{
    __shared__ float redA[256], redB[256];
    const int tid = threadIdx.x;
    const int c = tid >> 4, j = tid & 15;
    float s0 = 0.f, q0 = 0.f;
    for (int b = j; b < BATCH; b += 16) {
        s0 += bs[b * 32 + c];
        q0 += bs[b * 32 + 16 + c];
    }
    redA[tid] = s0; redB[tid] = q0;
    __syncthreads();
    for (int off = 8; off > 0; off >>= 1) {
        if (j < off) { redA[tid] += redA[tid + off]; redB[tid] += redB[tid + off]; }
        __syncthreads();
    }
    if (j == 0) {
        float mean = redA[tid] * invN;
        float var  = redB[tid] * invN - mean * mean;
        float sc   = g[c] * rsqrtf(var + EPS);
        st[c]      = sc;
        st[16 + c] = bb[c] - sc * mean;
    }
}

// ---------------------------------------------------------------------------
// Kernel 5: apply BN1+relu -> h1 (f16), channel means -> attention-2 ->
// per-sample conv2 weights w2h (f16, K-major K = tap*16 + in_ch).
// ---------------------------------------------------------------------------
__global__ void __launch_bounds__(256) k_pass2(
    const float* __restrict__ out1, const float* __restrict__ st,
    const float* __restrict__ a2w1, const float* __restrict__ a2b1,
    const float* __restrict__ a2w2, const float* __restrict__ a2b2,
    const float* __restrict__ w2k,
    _Float16* __restrict__ h1, _Float16* __restrict__ w2h)
{
    __shared__ float redA[256];
    __shared__ float p2s[FCH];
    __shared__ float attS[K4];
    const int tid = threadIdx.x;
    const int b   = blockIdx.x;
    const int c = tid >> 4, j = tid & 15;
    const float a  = st[c];
    const float sh = st[16 + c];
    float sum = 0.f;
    for (int t = j; t < NF; t += 16) {
        const size_t idx = (size_t)b * (FCH * NF) + c * NF + t;
        float y = fmaxf(a * out1[idx] + sh, 0.f);
        h1[idx] = (_Float16)y;
        sum += y;
    }
    redA[tid] = sum;
    __syncthreads();
    for (int off = 8; off > 0; off >>= 1) {
        if (j < off) redA[tid] += redA[tid + off];
        __syncthreads();
    }
    if (j == 0) p2s[c] = redA[tid] / (float)NF;
    __syncthreads();
    if (tid == 0) {
        float hh[4];
        for (int r = 0; r < 4; ++r) {
            float acc = a2b1[r];
            for (int cc = 0; cc < FCH; ++cc) acc += p2s[cc] * a2w1[r * FCH + cc];
            hh[r] = fmaxf(acc, 0.f);
        }
        float lg[K4]; float mx = -1e30f;
        for (int k = 0; k < K4; ++k) {
            float acc = a2b2[k];
            for (int r = 0; r < 4; ++r) acc += hh[r] * a2w2[k * 4 + r];
            lg[k] = acc * (1.0f / TEMP);
            mx = fmaxf(mx, lg[k]);
        }
        float ssum = 0.f;
        for (int k = 0; k < K4; ++k) { lg[k] = __expf(lg[k] - mx); ssum += lg[k]; }
        for (int k = 0; k < K4; ++k) attS[k] = lg[k] / ssum;
    }
    __syncthreads();
    // w2h[b][o*160 + tap*16 + i] = sum_k att[k] * w2k[k,o,i,0,tap]   (K-major f16)
    for (int idx = tid; idx < FCH * FCH * 10; idx += 256) {
        const int o   = idx / 160;
        const int rem = idx - o * 160;
        const int kk  = rem >> 4;     // tap
        const int i   = rem & 15;     // in channel
        float acc = 0.f;
        for (int k = 0; k < K4; ++k)
            acc += attS[k] * w2k[k * 2560 + o * 160 + i * 10 + kk];
        w2h[(size_t)b * 2560 + idx] = (_Float16)acc;
    }
}

// ---------------------------------------------------------------------------
// Kernel 6: dynamic conv2 (dilated, 10 taps) via f16 WMMA.  M=16 out-ch,
// N=405 (26 tiles, last padded), K=160 (5 steps of 32).  A (per-sample
// weights) and B source (h1[b]) staged in LDS via async DMA.
// ---------------------------------------------------------------------------
__global__ void __launch_bounds__(256) k_conv2(
    const _Float16* __restrict__ h1, const _Float16* __restrict__ w2h,
    float* __restrict__ out2, float* __restrict__ bs2)
{
    __shared__ _Float16 w2s[FCH * 160];   // 2560 halves
    __shared__ _Float16 h1s[FCH * NF];    // 6912 halves
    __shared__ float outS[FCH * 416];     // padded time dim
    __shared__ float redA[256], redB[256];
    const int tid  = threadIdx.x;
    const int b    = blockIdx.x;
    const int lane = tid & 31;
    const int wave = tid >> 5;
    const int half_ = lane >> 4;
    const int nloc  = lane & 15;

#if USE_ASYNC_LDS
    {
        const _Float16* gw = w2h + (size_t)b * 2560;
        const _Float16* gh = h1 + (size_t)b * (FCH * NF);
        for (int i = tid; i < 2560 / 8; i += 256)      // 320 x 16B chunks
            __builtin_amdgcn_global_load_async_to_lds_b128(
                (i4*)(gw + i * 8), (i4*)(w2s + i * 8), 0, 0);
        for (int i = tid; i < (FCH * NF) / 8; i += 256) // 864 x 16B chunks
            __builtin_amdgcn_global_load_async_to_lds_b128(
                (i4*)(gh + i * 8), (i4*)(h1s + i * 8), 0, 0);
        __builtin_amdgcn_s_wait_asynccnt(0);
    }
#else
    for (int i = tid; i < FCH * 160; i += 256) w2s[i] = w2h[(size_t)b * 2560 + i];
    for (int i = tid; i < FCH * NF; i += 256)  h1s[i] = h1[(size_t)b * (FCH * NF) + i];
#endif
    __syncthreads();

    // A fragments from LDS weights (K-major already)
    v16h Af[5];
    {
        const int o     = nloc;
        const int kbase = half_ * 16;
#pragma unroll
        for (int s = 0; s < 5; ++s) {
            v16h av;
#pragma unroll
            for (int e = 0; e < 16; ++e) {
                const int K = s * 32 + kbase + e;
                av[e] = w2s[o * 160 + K];
            }
            Af[s] = av;
        }
    }

    const int kb2 = half_ * 16;
    for (int tile = wave; tile < 26; tile += 8) {
        const int tt = tile * 16 + nloc;
        const int tc = tt < OUTW ? tt : (OUTW - 1);   // clamp padded lanes
        v8f acc = {};
#pragma unroll
        for (int s = 0; s < 5; ++s) {
            v16h bf;
#pragma unroll
            for (int e = 0; e < 16; ++e) {
                const int K = s * 32 + kb2 + e;       // K = tap*16 + in_ch
                bf[e] = h1s[(K & 15) * NF + tc + 3 * (K >> 4)];
            }
            acc = __builtin_amdgcn_wmma_f32_16x16x32_f16(
                false, Af[s], false, bf, (short)0, acc, false, false);
        }
#pragma unroll
        for (int r = 0; r < 8; ++r) {
            const int o = r + 8 * half_;
            outS[o * 416 + tt] = acc[r];
            if (tt < OUTW) out2[(size_t)b * (FCH * OUTW) + o * OUTW + tt] = acc[r];
        }
    }
    __syncthreads();

    const int c = tid >> 4, j = tid & 15;
    float s0 = 0.f, q0 = 0.f;
    for (int t = j; t < OUTW; t += 16) {
        float v = outS[c * 416 + t];
        s0 += v; q0 += v * v;
    }
    redA[tid] = s0; redB[tid] = q0;
    __syncthreads();
    for (int off = 8; off > 0; off >>= 1) {
        if (j < off) { redA[tid] += redA[tid + off]; redB[tid] += redB[tid + off]; }
        __syncthreads();
    }
    if (j == 0) {
        bs2[b * 32 + c]      = redA[tid];
        bs2[b * 32 + 16 + c] = redB[tid];
    }
}

// ---------------------------------------------------------------------------
// Kernel 8: fused BN2+relu + FC (N=4).  One block per sample.
// ---------------------------------------------------------------------------
__global__ void __launch_bounds__(256) k_fc(
    const float* __restrict__ out2, const float* __restrict__ st2,
    const float* __restrict__ fcw, const float* __restrict__ fcb,
    float* __restrict__ outp)
{
    __shared__ float a2s[FCH], s2s[FCH];
    __shared__ float redA[256];
    const int tid = threadIdx.x;
    const int b   = blockIdx.x;
    if (tid < FCH) a2s[tid] = st2[tid];
    else if (tid < 2 * FCH) s2s[tid - FCH] = st2[tid];
    __syncthreads();
    float acc[NCLS] = {0.f, 0.f, 0.f, 0.f};
    for (int idx = tid; idx < FCH * OUTW; idx += 256) {
        const int o = idx / OUTW;
        float y = fmaxf(a2s[o] * out2[(size_t)b * (FCH * OUTW) + idx] + s2s[o], 0.f);
#pragma unroll
        for (int n = 0; n < NCLS; ++n) acc[n] += y * fcw[n * (FCH * OUTW) + idx];
    }
    for (int n = 0; n < NCLS; ++n) {
        redA[tid] = acc[n];
        __syncthreads();
        for (int off = 128; off > 0; off >>= 1) {
            if (tid < off) redA[tid] += redA[tid + off];
            __syncthreads();
        }
        if (tid == 0) outp[b * NCLS + n] = redA[0] + fcb[n];
        __syncthreads();
    }
}

// ---------------------------------------------------------------------------
// Workspace layout (bytes, all 256-aligned; buffers aliased where lifetimes
// do not overlap):
//   [0,            56,623,104)  Xf f32 (DFT out)  -- later reused as h1 f16
//   [56,623,104,   57,065,472)  tab f16 (432x512)
//   [57,065,472,  170,311,680)  out1 f32          -- later reused as out2 f32
//   [170,311,680, 191,283,200)  w2h f16 (4096x2560)
//   [191,283,200, 191,807,488)  bs1 (4096x32 f32)
//   [191,807,488, 192,331,776)  bs2 (4096x32 f32)
//   [192,331,776, 192,332,032)  stats: a1/sh1 (32), a2/sh2 (32)
// Total ~183.4 MiB.
// ---------------------------------------------------------------------------
extern "C" void kernel_launch(void* const* d_in, const int* in_sizes, int n_in,
                              void* d_out, int out_size, void* d_ws, size_t ws_size,
                              hipStream_t stream)
{
    const float* x    = (const float*)d_in[0];
    const float* w1k  = (const float*)d_in[1];
    const float* a1w1 = (const float*)d_in[2];
    const float* a1b1 = (const float*)d_in[3];
    const float* a1w2 = (const float*)d_in[4];
    const float* a1b2 = (const float*)d_in[5];
    const float* g1   = (const float*)d_in[6];
    const float* b1   = (const float*)d_in[7];
    const float* w2k  = (const float*)d_in[8];
    const float* a2w1 = (const float*)d_in[9];
    const float* a2b1 = (const float*)d_in[10];
    const float* a2w2 = (const float*)d_in[11];
    const float* a2b2 = (const float*)d_in[12];
    const float* g2   = (const float*)d_in[13];
    const float* b2   = (const float*)d_in[14];
    const float* fcw  = (const float*)d_in[15];
    const float* fcb  = (const float*)d_in[16];
    float* outp = (float*)d_out;

    char* w = (char*)d_ws;
    float*    Xf   = (float*)(w + 0);
    _Float16* h1   = (_Float16*)(w + 0);               // aliases Xf (Xf dead)
    _Float16* tab  = (_Float16*)(w + 56623104);
    float*    out1 = (float*)(w + 57065472);
    float*    out2 = (float*)(w + 57065472);           // aliases out1 (dead)
    _Float16* w2h  = (_Float16*)(w + 170311680);
    float*    bs1  = (float*)(w + 191283200);
    float*    bs2  = (float*)(w + 191807488);
    float*    st   = (float*)(w + 192331776);          // [0..31]=bn1, [32..63]=bn2

    k_build_tab<<<(NF * TLEN) / 256, 256, 0, stream>>>(tab);
    k_dft_gemm<<<(BATCH * CHANS) / 128, 256, 0, stream>>>(x, tab, Xf);
    k_conv1<<<BATCH, 256, 0, stream>>>(Xf, w1k, a1w1, a1b1, a1w2, a1b2, out1, bs1);
    k_bn_reduce<<<1, 256, 0, stream>>>(bs1, g1, b1, st,
                                       1.0f / (float)((size_t)BATCH * NF));
    k_pass2<<<BATCH, 256, 0, stream>>>(out1, st, a2w1, a2b1, a2w2, a2b2, w2k, h1, w2h);
    k_conv2<<<BATCH, 256, 0, stream>>>(h1, w2h, out2, bs2);
    k_bn_reduce<<<1, 256, 0, stream>>>(bs2, g2, b2, st + 32,
                                       1.0f / (float)((size_t)BATCH * OUTW));
    k_fc<<<BATCH, 256, 0, stream>>>(out2, st + 32, fcw, fcb, outp);
}
